// GeomolGNNOGBFeat_36773509989186
// MI455X (gfx1250) — compile-verified, matrix operand order checked by reading, not run on hardware
//
#include <hip/hip_runtime.h>
#include <hip/hip_bf16.h>

// GeoMol GNN (OGB features) for MI455X / gfx1250.
// All Linear layers run as bf16 WMMA GEMMs (v_wmma_f32_16x16x32_bf16, f32
// accumulate). H=300 padded to 320. Concat inputs = multi-segment GEMMs with
// fused row-index gather. A tiles are staged to LDS via the CDNA5 async
// global->LDS path (global_load_async_to_lds_b128 + s_wait_asynccnt), with
// padded LDS stride to avoid bank conflicts. BatchNorm column stats are fused
// into the GEMM epilogue (shuffle + atomics), removing a full re-read of C.
// Linear bias is skipped: training-mode BN subtracts the batch mean, which
// cancels any constant bias exactly.

#define NN 50000
#define EE 256000
#define HH 300
#define HP 320
#define BN_EPS 1e-5f

#define TILE_R 128
#define TILE_N 64
#define LDSS 328   // padded LDS row stride (elements); 328*2B=656B -> conflict-free

typedef __attribute__((ext_vector_type(8)))  __bf16 v8bf;
typedef __attribute__((ext_vector_type(16))) __bf16 v16bf;
typedef __attribute__((ext_vector_type(8)))  float  v8f;

__device__ __constant__ int c_atom_off[9] = {0,119,123,135,147,157,163,169,171};
__device__ __constant__ int c_bond_off[3] = {0,5,11};

__device__ __forceinline__ unsigned short f2bf(float f) {
    unsigned int u = __float_as_uint(f);
    unsigned int r = u + 0x7FFFu + ((u >> 16) & 1u);   // round-to-nearest-even
    return (unsigned short)(r >> 16);
}
__device__ __forceinline__ float bf2f(unsigned short h) {
    return __uint_as_float(((unsigned int)h) << 16);
}

// CDNA5 async global->LDS copy, 16B per lane. %0 = LDS byte offset (VGPR),
// %1 = 64-bit global address (VGPR pair). Tracked by ASYNCcnt.
__device__ __forceinline__ void async_ld16(unsigned lds_off, const void* gaddr) {
    asm volatile("global_load_async_to_lds_b128 %0, %1, off"
                 :: "v"(lds_off), "v"(gaddr) : "memory");
}
__device__ __forceinline__ void wait_async0() {
    asm volatile("s_wait_asynccnt 0x0" ::: "memory");
}

// ---------------------------------------------------------------------------
// small utility kernels
// ---------------------------------------------------------------------------
__global__ void zero_f32(float* p, long n) {
    long i = (long)blockIdx.x * blockDim.x + threadIdx.x;
    if (i < n) p[i] = 0.0f;
}

__global__ void count_edges(const int* col, float* cnt) {
    long e = (long)blockIdx.x * blockDim.x + threadIdx.x;
    if (e < EE) atomicAdd(&cnt[col[e]], 1.0f);
}

// Wt[n][k] = W[k][n] in bf16, padded to [320,320] with zeros.
__global__ void prep_w(const float* __restrict__ W, int Ksrc, unsigned short* __restrict__ Wt) {
    int i = blockIdx.x * blockDim.x + threadIdx.x;
    if (i >= HP * HP) return;
    int n = i / HP, k = i % HP;
    Wt[i] = (n < HH && k < Ksrc) ? f2bf(W[(long)k * HH + n]) : (unsigned short)0;
}

// node encoder: sum of 9 atom embeddings | rand_x | zero pad   -> bf16 [N,320]
__global__ void embed_nodes(const int* __restrict__ x, const float* __restrict__ rx,
                            const float* __restrict__ aemb, unsigned short* __restrict__ A) {
    long i = (long)blockIdx.x * blockDim.x + threadIdx.x;
    if (i >= (long)NN * HP) return;
    int c = (int)(i % HP); long n = i / HP;
    float v = 0.0f;
    if (c < HH) {
        #pragma unroll
        for (int j = 0; j < 9; ++j)
            v += aemb[(long)(x[n * 9 + j] + c_atom_off[j]) * HH + c];
    } else if (c < HH + 10) {
        v = rx[n * 10 + (c - HH)];
    }
    A[i] = (c < HH + 10) ? f2bf(v) : (unsigned short)0;
}

__global__ void embed_edges(const int* __restrict__ ea, const float* __restrict__ re,
                            const float* __restrict__ bemb, unsigned short* __restrict__ A) {
    long i = (long)blockIdx.x * blockDim.x + threadIdx.x;
    if (i >= (long)EE * HP) return;
    int c = (int)(i % HP); long e = i / HP;
    float v = 0.0f;
    if (c < HH) {
        #pragma unroll
        for (int j = 0; j < 3; ++j)
            v += bemb[(long)(ea[e * 3 + j] + c_bond_off[j]) * HH + c];
    } else if (c < HH + 10) {
        v = re[e * 10 + (c - HH)];
    }
    A[i] = (c < HH + 10) ? f2bf(v) : (unsigned short)0;
}

// ---------------------------------------------------------------------------
// WMMA GEMM:  C[rows,320] = sum_seg  A_seg[idx_seg(r), 0:320] @ Wt_seg^T
//   A: bf16 row-major, lda=320 (staged to LDS, per-row gather, async copies).
//   Wt: bf16 [320(n), 320(k)] row-major (B^T), read from global (L2 resident).
// Block: 256 thr = 8 waves; block tile 128 rows x 64 cols; wave tile 64x16.
// Epilogue: C store + fused per-column BatchNorm partial sums into `stats`.
// ---------------------------------------------------------------------------
struct GemmSeg { const unsigned short* A; const int* idx; const unsigned short* W; };

__device__ __forceinline__ v16bf load_frag(const unsigned short* base, int koff) {
    v8bf lo = *reinterpret_cast<const v8bf*>(base + koff);        // K = koff..koff+7
    v8bf hi = *reinterpret_cast<const v8bf*>(base + koff + 16);   // K = koff+16..+23
    return __builtin_shufflevector(lo, hi, 0,1,2,3,4,5,6,7,8,9,10,11,12,13,14,15);
}

__global__ void __launch_bounds__(256)
gemm_bf16(GemmSeg s0, GemmSeg s1, GemmSeg s2, int nseg,
          float* __restrict__ C, float* __restrict__ stats, int rows) {
    __shared__ unsigned short sA[TILE_R * LDSS];   // 84 KB of 320 KB WGP LDS

    const int tid  = threadIdx.x;
    const int lane = tid & 31;
    const int w    = tid >> 5;
    const int rowBase0 = blockIdx.x * TILE_R;
    const int nBase0   = blockIdx.y * TILE_N;
    const int waveRow  = (w >> 2) * 64;   // wave's row offset inside tile
    const int waveN    = (w & 3) * 16;    // wave's col offset inside tile

    const int half8 = ((lane >> 4) & 1) * 8;   // lanes 16-31 take K offset +8
    const int ml    = lane & 15;

    // staging map: 2 threads per A row, each copies half a row (320 B = 20x16B)
    const int a_row  = tid >> 1;
    const int a_half = tid & 1;

    v8f acc[4] = {};
    GemmSeg segs[3] = {s0, s1, s2};

    #pragma unroll
    for (int s = 0; s < 3; ++s) {
        if (s >= nseg) break;
        GemmSeg sg = segs[s];

        // ---- stage A tile into LDS via async copies (fused gather) ----
        {
            int r = rowBase0 + a_row;
            if (r > rows - 1) r = rows - 1;
            int ar = sg.idx ? sg.idx[r] : r;
            const unsigned short* g = sg.A + (long)ar * HP + a_half * 160;
            unsigned lds = (unsigned)(size_t)&sA[a_row * LDSS + a_half * 160];
            #pragma unroll
            for (int j = 0; j < 20; ++j)
                async_ld16(lds + j * 16, g + j * 8);
        }
        wait_async0();
        __syncthreads();

        const unsigned short* Ab[4];
        #pragma unroll
        for (int t = 0; t < 4; ++t)
            Ab[t] = &sA[(waveRow + t * 16 + ml) * LDSS];
        const unsigned short* Bb = sg.W + (long)(nBase0 + waveN + ml) * HP;

        #pragma unroll
        for (int k0 = 0; k0 < HP; k0 += 32) {
            v16bf b = load_frag(Bb, k0 + half8);
            #pragma unroll
            for (int t = 0; t < 4; ++t) {
                v16bf a = load_frag(Ab[t], k0 + half8);
                acc[t] = __builtin_amdgcn_wmma_f32_16x16x32_bf16(
                    false, a, false, b, (short)0, acc[t], false, false);
            }
        }
        __syncthreads();   // protect sA before next segment overwrites it
    }

    // ---- epilogue: store C + fused BN partial stats ----
    const int half = (lane >> 4) & 1;
    const int n = nBase0 + waveN + ml;
    float ps = 0.0f, ps2 = 0.0f;
    #pragma unroll
    for (int t = 0; t < 4; ++t) {
        #pragma unroll
        for (int v = 0; v < 8; ++v) {
            int m = rowBase0 + waveRow + t * 16 + v + 8 * half;
            if (m < rows) {
                float x = acc[t][v];
                C[(long)m * HP + n] = x;
                ps += x; ps2 += x * x;
            }
        }
    }
    // combine the two half-wave lanes that own the same column n
    ps  += __shfl_xor(ps, 16);
    ps2 += __shfl_xor(ps2, 16);
    if (lane < 16 && n < HH) {
        atomicAdd(&stats[n],      ps);
        atomicAdd(&stats[HP + n], ps2);
    }
}

// ---------------------------------------------------------------------------
// BatchNorm (training-mode batch stats) + ReLU, bf16 requant (+optional f32 out)
// ---------------------------------------------------------------------------
__global__ void bn_apply(const float* __restrict__ C, const float* __restrict__ stats,
                         const float* __restrict__ g, const float* __restrict__ be,
                         int rows, unsigned short* __restrict__ out_bf,
                         float* __restrict__ out_f32) {
    long i = (long)blockIdx.x * blockDim.x + threadIdx.x;
    if (i >= (long)rows * HP) return;
    int c = (int)(i % HP); long r = i / HP;
    if (c < HH) {
        float inv = 1.0f / (float)rows;
        float mu  = stats[c] * inv;
        float var = stats[HP + c] * inv - mu * mu;
        float v = (C[i] - mu) * rsqrtf(var + BN_EPS) * g[c] + be[c];
        v = fmaxf(v, 0.0f);
        out_bf[i] = f2bf(v);
        if (out_f32) out_f32[r * HH + c] = v;
    } else {
        out_bf[i] = 0;
    }
}

// ---------------------------------------------------------------------------
// scatter-mean aggregation
// ---------------------------------------------------------------------------
__global__ void scatter_add(const unsigned short* __restrict__ m, const int* __restrict__ col,
                            float* __restrict__ agg) {
    long i = (long)blockIdx.x * blockDim.x + threadIdx.x;
    if (i >= (long)EE * HH) return;
    int c = (int)(i % HH); long e = i / HH;
    atomicAdd(&agg[(long)col[e] * HP + c], bf2f(m[e * HP + c]));
}

__global__ void agg_finalize(const float* __restrict__ agg, const float* __restrict__ cnt,
                             unsigned short* __restrict__ out_bf) {
    long i = (long)blockIdx.x * blockDim.x + threadIdx.x;
    if (i >= (long)NN * HP) return;
    int c = (int)(i % HP); long n = i / HP;
    if (c < HH) {
        float d = cnt[n]; d = (d > 1.0f) ? d : 1.0f;
        out_bf[i] = f2bf(agg[i] / d);
    } else {
        out_bf[i] = 0;
    }
}

// ---------------------------------------------------------------------------
// host orchestration
// ---------------------------------------------------------------------------
extern "C" void kernel_launch(void* const* d_in, const int* in_sizes, int n_in,
                              void* d_out, int out_size, void* d_ws, size_t ws_size,
                              hipStream_t stream) {
    // d_in layout (setup_inputs insertion order, params flattened in dict order,
    // each linear layer as W,b,g,be):
    const int*   x     = (const int*)  d_in[0];
    const int*   ei    = (const int*)  d_in[1];
    const int*   ea    = (const int*)  d_in[2];
    const float* rx    = (const float*)d_in[3];
    const float* re    = (const float*)d_in[4];
    const float* aemb  = (const float*)d_in[5];
    const float* bemb  = (const float*)d_in[6];
    const int* row = ei;
    const int* col = ei + EE;

    // workspace carve-out
    size_t off = 0;
    auto give = [&](size_t bytes) -> void* {
        void* p = (char*)d_ws + off;
        off += (bytes + 255) & ~(size_t)255;
        return p;
    };
    unsigned short* wt     = (unsigned short*)give((size_t)12 * HP * HP * 2);
    unsigned short* h_bf   = (unsigned short*)give((size_t)NN * HP * 2);
    unsigned short* e_bf   = (unsigned short*)give((size_t)EE * HP * 2);
    unsigned short* t_bf   = (unsigned short*)give((size_t)EE * HP * 2);  // + A-edge staging
    unsigned short* m_bf   = (unsigned short*)give((size_t)EE * HP * 2);  // + A-node staging
    float*          Cbuf   = (float*)         give((size_t)EE * HP * 4);
    float*          agg_f  = (float*)         give((size_t)NN * HP * 4);
    unsigned short* agg_bf = (unsigned short*)give((size_t)NN * HP * 2);
    float*          cnt    = (float*)         give((size_t)NN * 4);
    float*          stats  = (float*)         give((size_t)2 * HP * 4);
    (void)ws_size; (void)in_sizes; (void)n_in; (void)out_size;

    float* out_h = (float*)d_out;            // [N,300]
    float* out_e = out_h + (size_t)NN * HH;  // [E,300]

    auto WT = [&](int i) { return wt + (size_t)i * HP * HP; };

    // --- weight prep: 12 transposed bf16 blocks ---
    struct { const float* src; int Ksrc; } wdesc[12] = {
        { (const float*)d_in[7],               310 },  // node_init
        { (const float*)d_in[11],              310 },  // edge_init
        { (const float*)d_in[15],              300 },  // edge_model L0 (h[row])
        { (const float*)d_in[15] + 300 * HH,   300 },  // edge_model L0 (h[col])
        { (const float*)d_in[15] + 600 * HH,   300 },  // edge_model L0 (e)
        { (const float*)d_in[19],              300 },  // edge_model L1
        { (const float*)d_in[23],              300 },  // node_mlp1 L0 (h[row])
        { (const float*)d_in[23] + 300 * HH,   300 },  // node_mlp1 L0 (e)
        { (const float*)d_in[27],              300 },  // node_mlp1 L1
        { (const float*)d_in[31],              300 },  // node_mlp2 L0 (h)
        { (const float*)d_in[31] + 300 * HH,   300 },  // node_mlp2 L0 (agg)
        { (const float*)d_in[35],              300 },  // node_mlp2 L1
    };
    for (int i = 0; i < 12; ++i)
        prep_w<<<(HP * HP + 255) / 256, 256, 0, stream>>>(wdesc[i].src, wdesc[i].Ksrc, WT(i));

    // --- degree counts ---
    zero_f32<<<(NN + 255) / 256, 256, 0, stream>>>(cnt, NN);
    count_edges<<<(EE + 255) / 256, 256, 0, stream>>>(col, cnt);

    // --- one Linear+BN+ReLU layer (stats fused into GEMM epilogue) ---
    auto run_layer = [&](int nseg, GemmSeg s0, GemmSeg s1, GemmSeg s2, int rows,
                         const float* g, const float* be,
                         unsigned short* out_bf, float* out_f32) {
        zero_f32<<<(2 * HP + 255) / 256, 256, 0, stream>>>(stats, 2 * HP);
        dim3 grid((rows + TILE_R - 1) / TILE_R, HP / TILE_N);
        gemm_bf16<<<grid, 256, 0, stream>>>(s0, s1, s2, nseg, Cbuf, stats, rows);
        long tot = (long)rows * HP;
        bn_apply<<<(tot + 255) / 256, 256, 0, stream>>>(Cbuf, stats, g, be, rows, out_bf, out_f32);
    };
    GemmSeg Z = { nullptr, nullptr, nullptr };

    // --- encoders + init MLPs ---
    unsigned short* Anode = m_bf;   // staging (free until node_mlp1 runs)
    unsigned short* Aedge = t_bf;   // staging (free until edge_model runs)
    {
        long tn = (long)NN * HP, te = (long)EE * HP;
        embed_nodes<<<(tn + 255) / 256, 256, 0, stream>>>(x, rx, aemb, Anode);
        embed_edges<<<(te + 255) / 256, 256, 0, stream>>>(ea, re, bemb, Aedge);
    }
    run_layer(1, {Anode, nullptr, WT(0)}, Z, Z, NN,
              (const float*)d_in[9],  (const float*)d_in[10], h_bf, nullptr);
    run_layer(1, {Aedge, nullptr, WT(1)}, Z, Z, EE,
              (const float*)d_in[13], (const float*)d_in[14], e_bf, nullptr);

    // --- message-passing depth loop ---
    for (int d = 0; d < 3; ++d) {
        bool last = (d == 2);
        // EdgeModel: e = MLP([h[row], h[col], e])
        run_layer(3, {h_bf, row, WT(2)}, {h_bf, col, WT(3)}, {e_bf, nullptr, WT(4)}, EE,
                  (const float*)d_in[17], (const float*)d_in[18], t_bf, nullptr);
        run_layer(1, {t_bf, nullptr, WT(5)}, Z, Z, EE,
                  (const float*)d_in[21], (const float*)d_in[22], e_bf,
                  last ? out_e : nullptr);
        // NodeModel part 1: m = MLP([h[row], e])
        run_layer(2, {h_bf, row, WT(6)}, {e_bf, nullptr, WT(7)}, Z, EE,
                  (const float*)d_in[25], (const float*)d_in[26], t_bf, nullptr);
        run_layer(1, {t_bf, nullptr, WT(8)}, Z, Z, EE,
                  (const float*)d_in[29], (const float*)d_in[30], m_bf, nullptr);
        // scatter-mean by col
        {
            long tn = (long)NN * HP, ts = (long)EE * HH;
            zero_f32<<<(tn + 255) / 256, 256, 0, stream>>>(agg_f, tn);
            scatter_add<<<(ts + 255) / 256, 256, 0, stream>>>(m_bf, col, agg_f);
            agg_finalize<<<(tn + 255) / 256, 256, 0, stream>>>(agg_f, cnt, agg_bf);
        }
        // NodeModel part 2: h = MLP([h, agg])
        run_layer(2, {h_bf, nullptr, WT(9)}, {agg_bf, nullptr, WT(10)}, Z, NN,
                  (const float*)d_in[33], (const float*)d_in[34], t_bf, nullptr);
        run_layer(1, {t_bf, nullptr, WT(11)}, Z, Z, NN,
                  (const float*)d_in[37], (const float*)d_in[38], h_bf,
                  last ? out_h : nullptr);
    }
}
